// Attention_62294205662064
// MI455X (gfx1250) — matrix operand order repaired and checked by reading, not compiled, not run on hardware
//
#include <hip/hip_runtime.h>
#include <cmath>

// ---------------- problem constants (match reference) ----------------
constexpr int BATCH = 2;
constexpr int SEQ   = 2048;
constexpr int EMB   = 2048;
constexpr int NH    = 16;
constexpr int NKV   = 8;
constexpr int HD    = 128;
constexpr int BT    = BATCH * SEQ;          // 4096 rows
constexpr float ATTN_SCALE = 0.08838834764831845f;  // 1/sqrt(128)

typedef __attribute__((ext_vector_type(16))) __bf16 v16bf;
typedef __attribute__((ext_vector_type(8)))  float  v8f;

#define NEG_INF (-__builtin_inff())

// ---------------- CDNA5 async global -> LDS copy (16 bytes / lane) ----------------
// GLOBAL_LOAD_ASYNC_TO_LDS_B128: copies 16 bytes from global address vaddr into LDS at vdst; tracked by ASYNCcnt.
// LDS byte address = low 32 bits of the generic shared-memory pointer.
__device__ inline void async_cp16(unsigned lds, const void* g) {
    asm volatile("global_load_async_to_lds_b128 %0, %1, off"
                 :: "v"(lds), "v"((unsigned long long)(uintptr_t)g)
                 : "memory");
}
__device__ inline void wait_async0() {
    asm volatile("s_wait_asynccnt 0x0" ::: "memory");
}

// ---------------- WMMA fragment loaders (CDNA5 16-bit layouts, vectorized) -------
// A matrix 16x32 (MxK), row-major source, leading dim ld (elements, multiple of 8).
// Per lane: elements 0..7 -> k = kh*8+0..7 ; 8..15 -> k = 16+kh*8+0..7  => 2x b128.
__device__ inline v16bf load_A(const __bf16* __restrict__ src, int ld) {
    const int lane = threadIdx.x & 31;
    const int m = lane & 15, kh = lane >> 4;
    union { v16bf v; uint4 q[2]; } u;
    const __bf16* p = src + (size_t)m * ld + kh * 8;
    u.q[0] = *(const uint4*)(p);
    u.q[1] = *(const uint4*)(p + 16);
    return u.v;
}

// B matrix 32x16 (KxN) with source holding B^T row-major (N rows x K cols, ld row
// stride). Per lane: elements 0..7 -> k = kh*8+0..7 ; 8..15 -> k = 16+kh*8+0..7 => 2x b128.
__device__ inline v16bf load_BT(const __bf16* __restrict__ src, int ld) {
    const int lane = threadIdx.x & 31;
    const int n = lane & 15, kh = lane >> 4;
    union { v16bf v; uint4 q[2]; } u;
    const __bf16* p = src + (size_t)n * ld + kh * 8;
    u.q[0] = *(const uint4*)(p);
    u.q[1] = *(const uint4*)(p + 16);
    return u.v;
}

__device__ inline v8f wmma_bf16(v16bf a, v16bf b, v8f c) {
    return __builtin_amdgcn_wmma_f32_16x16x32_bf16(false, a, false, b, (short)0, c,
                                                   false, false);
}

// reductions across the 16 lanes holding one matrix row (xor<16 stays in half-wave)
__device__ inline float hmax16(float x) {
    x = fmaxf(x, __shfl_xor(x, 1, 32));
    x = fmaxf(x, __shfl_xor(x, 2, 32));
    x = fmaxf(x, __shfl_xor(x, 4, 32));
    x = fmaxf(x, __shfl_xor(x, 8, 32));
    return x;
}
__device__ inline float hsum16(float x) {
    x += __shfl_xor(x, 1, 32);
    x += __shfl_xor(x, 2, 32);
    x += __shfl_xor(x, 4, 32);
    x += __shfl_xor(x, 8, 32);
    return x;
}

// ---------------- elementwise cast ----------------
__global__ void cast_f32_bf16(const float* __restrict__ src, __bf16* __restrict__ dst, int n) {
    int i = blockIdx.x * blockDim.x + threadIdx.x;
    int stride = gridDim.x * blockDim.x;
    for (; i < n; i += stride) dst[i] = (__bf16)src[i];
}

// ---------------- tiled transpose + cast: src f32 (R x C) -> dst bf16 (C x R) -----
__global__ void transpose_cast(const float* __restrict__ src, __bf16* __restrict__ dst,
                               int R, int C) {
    __shared__ float tile[32][33];
    const int c0 = blockIdx.x * 32, r0 = blockIdx.y * 32;
    const int tx = threadIdx.x, ty = threadIdx.y;
#pragma unroll
    for (int i = 0; i < 32; i += 8)
        tile[ty + i][tx] = src[(size_t)(r0 + ty + i) * C + c0 + tx];
    __syncthreads();
#pragma unroll
    for (int i = 0; i < 32; i += 8)
        dst[(size_t)(c0 + ty + i) * R + r0 + tx] = (__bf16)tile[tx][ty + i];
}

// ---------------- V: (B,T,NKV,D) f32 -> (B,NKV,D,T) bf16 (transposed per head) ----
__global__ void v_transpose_cast(const float* __restrict__ vlin, __bf16* __restrict__ vT) {
    const int h = blockIdx.z % NKV, b = blockIdx.z / NKV;
    const int d0 = blockIdx.x * 32, t0 = blockIdx.y * 32;
    const float* src = vlin + ((size_t)b * SEQ * NKV + h) * HD;
    __bf16* dst = vT + ((size_t)b * NKV + h) * (size_t)HD * SEQ;
    __shared__ float tile[32][33];
    const int tx = threadIdx.x, ty = threadIdx.y;
#pragma unroll
    for (int i = 0; i < 32; i += 8)
        tile[ty + i][tx] = src[(size_t)(t0 + ty + i) * (NKV * HD) + d0 + tx];
    __syncthreads();
#pragma unroll
    for (int i = 0; i < 32; i += 8)
        dst[(size_t)(d0 + ty + i) * SEQ + t0 + tx] = (__bf16)tile[tx][ty + i];
}

// ---------------- bf16 GEMM: C(f32, MxN) = A(bf16, MxK) @ B (B given as B^T) ------
// Bt is (N x K) row-major. 128 threads (4 waves). Block tile 128(M) x 64(N), K-step
// 32, double-buffered LDS filled by async global->LDS copies. Wave: 32x64 sub-tile.
__global__ __launch_bounds__(128) void gemm_bf16_f32(const __bf16* __restrict__ A,
                                                     const __bf16* __restrict__ Bt,
                                                     float* __restrict__ C,
                                                     int M, int N, int K) {
    __shared__ __align__(16) __bf16 As[2][128 * 32];   // 8 KB per buffer
    __shared__ __align__(16) __bf16 BsT[2][64 * 32];   // 4 KB per buffer
    const int tid  = threadIdx.x;
    const int lane = tid & 31;
    const int wave = tid >> 5;
    const int rowBase = blockIdx.y * 128;
    const int colBase = blockIdx.x * 64;

    // fixed per-thread staging slots: A rows (tid>>2)+{0,32,64,96}, B^T rows
    // (tid>>2)+{0,32}, 16-byte column chunk (tid&3)*8.
    const __bf16* pA = A  + (size_t)(rowBase + (tid >> 2)) * K + (tid & 3) * 8;
    const __bf16* pB = Bt + (size_t)(colBase + (tid >> 2)) * K + (tid & 3) * 8;
    const size_t rstep = (size_t)32 * K;
    const unsigned lA = (unsigned)(uintptr_t)&As[0][0]  + tid * 16u;  // +buf*8192, +i*2048
    const unsigned lB = (unsigned)(uintptr_t)&BsT[0][0] + tid * 16u;  // +buf*4096, +i*2048

    auto stage = [&](int bufsel) {
        const unsigned a = lA + (unsigned)bufsel * 8192u;
        const unsigned b = lB + (unsigned)bufsel * 4096u;
        async_cp16(a,         pA);
        async_cp16(a + 2048u, pA + rstep);
        async_cp16(a + 4096u, pA + 2 * rstep);
        async_cp16(a + 6144u, pA + 3 * rstep);
        async_cp16(b,         pB);
        async_cp16(b + 2048u, pB + rstep);
        pA += 32; pB += 32;
    };

    stage(0);  // tile 0 -> buffer 0

    v8f acc0[4] = {}, acc1[4] = {};
    const int nkt = K / 32;
    for (int kt = 0; kt < nkt; ++kt) {
        const int buf = kt & 1;
        wait_async0();      // this wave's async fills done
        __syncthreads();    // all waves' fills of `buf` done
        if (kt + 1 < nkt) stage(buf ^ 1);  // fire next tile, overlaps the WMMAs

        const __bf16* aBase = As[buf] + wave * 32 * 32;
        v16bf fa0 = load_A(aBase, 32);
        v16bf fa1 = load_A(aBase + 16 * 32, 32);
        v16bf fb[4];
#pragma unroll
        for (int j = 0; j < 4; ++j) fb[j] = load_BT(BsT[buf] + j * 16 * 32, 32);
#pragma unroll
        for (int j = 0; j < 4; ++j) acc0[j] = wmma_bf16(fa0, fb[j], acc0[j]);
#pragma unroll
        for (int j = 0; j < 4; ++j) acc1[j] = wmma_bf16(fa1, fb[j], acc1[j]);
    }

    const int nloc = lane & 15, half = lane >> 4;
#pragma unroll
    for (int j = 0; j < 4; ++j)
#pragma unroll
        for (int r = 0; r < 8; ++r) {
            int m = r + half * 8;
            size_t row = (size_t)(rowBase + wave * 32 + m);
            C[row * N + colBase + j * 16 + nloc]        = acc0[j][r];
            C[(row + 16) * N + colBase + j * 16 + nloc] = acc1[j][r];
        }
}

// ---------------- RoPE + layout transpose: (B,T,nh,D) f32 -> (B,nh,T,D) bf16 -------
__global__ void rope_to_bf16(const float* __restrict__ lin, __bf16* __restrict__ out, int nh) {
    const int bid = blockIdx.x;
    const int t = bid % SEQ;
    const int h = (bid / SEQ) % nh;
    const int b = bid / (SEQ * nh);
    const int d = threadIdx.x;  // 0..63 (pair index)
    const float* src = lin + (((size_t)b * SEQ + t) * nh + h) * HD;
    __bf16* dst = out + (((size_t)b * nh + h) * SEQ + t) * HD;
    float inv = __expf(-(float)d * 0.14391156683f);  // 10000^(-d/64)
    float ang = (float)t * inv;
    float c = cosf(ang), s = sinf(ang);
    float q1 = src[2 * d], q2 = src[2 * d + 1];
    dst[d]      = (__bf16)(q1 * c - q2 * s);
    dst[64 + d] = (__bf16)(q2 * c + q1 * s);
}

// ---------------- flash attention (causal, GQA h -> h % NKV) -----------------------
// grid (SEQ/64, NH, BATCH), block 128. Wave w owns query rows [qblk*64+w*16, +16).
// K/V^T tiles double-buffered in LDS, filled with async global->LDS copies.
__global__ __launch_bounds__(128) void flash_attn(const __bf16* __restrict__ Qr,  // (B,NH,T,D)
                                                  const __bf16* __restrict__ Kr,  // (B,NKV,T,D)
                                                  const __bf16* __restrict__ Vt,  // (B,NKV,D,T)
                                                  __bf16* __restrict__ Out) {     // (B,T,NH*D)
    __shared__ __align__(16) __bf16 Ks[2][32 * 128];   // 8 KB per buffer
    __shared__ __align__(16) __bf16 VsT[2][128 * 32];  // 8 KB per buffer
    __shared__ __align__(16) __bf16 Ps[4][16 * 32];

    const int tid  = threadIdx.x;
    const int lane = tid & 31;
    const int wave = tid >> 5;
    const int qblk = blockIdx.x;
    const int h    = blockIdx.y;
    const int b    = blockIdx.z;
    const int hkv  = h % NKV;  // jnp.tile semantics

    const __bf16* qp = Qr + (((size_t)b * NH  + h)   * SEQ) * HD;
    const __bf16* kp = Kr + (((size_t)b * NKV + hkv) * SEQ) * HD;
    const __bf16* vp = Vt + (((size_t)b * NKV + hkv) * HD) * SEQ;
    const int qrow0 = qblk * 64 + wave * 16;

    v16bf qf[4];
#pragma unroll
    for (int kc = 0; kc < 4; ++kc) qf[kc] = load_A(qp + (size_t)qrow0 * HD + kc * 32, HD);

    // fixed per-thread staging slots:
    //   K tile: contiguous 8 KB -> slots (tid + i*128)*16 bytes
    //   V^T rows (tid>>2)+{0,32,64,96}, 16-byte chunk (tid&3)*8, row stride SEQ
    const __bf16* pK = kp + tid * 8;                                   // += 4096 / kb
    const __bf16* pV = vp + (size_t)(tid >> 2) * SEQ + (tid & 3) * 8;  // += 32 / kb
    const size_t vrow = (size_t)32 * SEQ;
    const unsigned lK = (unsigned)(uintptr_t)&Ks[0][0]  + tid * 16u;   // +buf*8192
    const unsigned lV = (unsigned)(uintptr_t)&VsT[0][0] + tid * 16u;   // +buf*8192

    auto stage = [&](int bufsel) {
        const unsigned k = lK + (unsigned)bufsel * 8192u;
        const unsigned v = lV + (unsigned)bufsel * 8192u;
        async_cp16(k,         pK);
        async_cp16(k + 2048u, pK + 1024);
        async_cp16(k + 4096u, pK + 2048);
        async_cp16(k + 6144u, pK + 3072);
        async_cp16(v,         pV);
        async_cp16(v + 2048u, pV + vrow);
        async_cp16(v + 4096u, pV + 2 * vrow);
        async_cp16(v + 6144u, pV + 3 * vrow);
        pK += 4096; pV += 32;
    };

    stage(0);  // tile kb=0 -> buffer 0

    v8f o[8] = {};
    float mrun[8], lrun[8];
#pragma unroll
    for (int r = 0; r < 8; ++r) { mrun[r] = NEG_INF; lrun[r] = 0.0f; }

    const int nloc = lane & 15, half = lane >> 4;
    const int nkb = qblk * 2 + 2;  // causal: keys [0, qblk*64+64)
    for (int kb = 0; kb < nkb; ++kb) {
        const int kbase = kb * 32;
        const int buf = kb & 1;
        wait_async0();      // this wave's async fills done
        __syncthreads();    // all waves' fills of `buf` done
        if (kb + 1 < nkb) stage(buf ^ 1);  // fire next tile during the WMMAs

        // S = Q (16x128) @ K^T (128x32) -> two 16x16 f32 tiles
        v8f s0 = {}, s1 = {};
#pragma unroll
        for (int kc = 0; kc < 4; ++kc) {
            v16bf b0 = load_BT(Ks[buf] + kc * 32, 128);
            v16bf b1 = load_BT(Ks[buf] + 16 * 128 + kc * 32, 128);
            s0 = wmma_bf16(qf[kc], b0, s0);
            s1 = wmma_bf16(qf[kc], b1, s1);
        }

        // online softmax (rows r / r+8 live in lane halves)
        __bf16* pw = &Ps[wave][0];
#pragma unroll
        for (int r = 0; r < 8; ++r) {
            int m  = r + half * 8;
            int qi = qrow0 + m;
            float x0 = s0[r] * ATTN_SCALE; if (kbase + nloc > qi)      x0 = NEG_INF;
            float x1 = s1[r] * ATTN_SCALE; if (kbase + 16 + nloc > qi) x1 = NEG_INF;
            float mx = hmax16(fmaxf(x0, x1));
            float mn = fmaxf(mrun[r], mx);
            float al = __expf(mrun[r] - mn);
            float p0 = __expf(x0 - mn);
            float p1 = __expf(x1 - mn);
            lrun[r] = lrun[r] * al + hsum16(p0 + p1);
            mrun[r] = mn;
#pragma unroll
            for (int dt = 0; dt < 8; ++dt) o[dt][r] *= al;
            pw[m * 32 + nloc]      = (__bf16)p0;
            pw[m * 32 + 16 + nloc] = (__bf16)p1;
        }
        // wave-local LDS RAW: P stored above, re-read below in A-fragment layout
        asm volatile("s_wait_dscnt 0" ::: "memory");

        v16bf pf = load_A(pw, 32);
#pragma unroll
        for (int dt = 0; dt < 8; ++dt) {
            v16bf bv = load_BT(VsT[buf] + dt * 16 * 32, 32);
            o[dt] = wmma_bf16(pf, bv, o[dt]);
        }
    }

    // normalize and store (B,T,NH*D) bf16 for the output projection GEMM
    __bf16* op = Out + ((size_t)b * SEQ) * (NH * HD) + (size_t)h * HD;
#pragma unroll
    for (int r = 0; r < 8; ++r) {
        int m = r + half * 8;
        int t = qrow0 + m;
        float inv = 1.0f / lrun[r];
#pragma unroll
        for (int dt = 0; dt < 8; ++dt)
            op[(size_t)t * (NH * HD) + dt * 16 + nloc] = (__bf16)(o[dt][r] * inv);
    }
}

// ---------------- host-side orchestration ----------------
extern "C" void kernel_launch(void* const* d_in, const int* in_sizes, int n_in,
                              void* d_out, int out_size, void* d_ws, size_t ws_size,
                              hipStream_t stream) {
    const float* x  = (const float*)d_in[0];
    // d_in[1] = mask (int32 causal tril) — causality computed analytically
    const float* wq = (const float*)d_in[2];
    const float* wk = (const float*)d_in[3];
    const float* wv = (const float*)d_in[4];
    const float* wo = (const float*)d_in[5];
    float* out = (float*)d_out;

    char* ws = (char*)d_ws;
    const size_t o_xb   = 0;
    const size_t o_wqT  = o_xb   + (size_t)BT * EMB * 2;
    const size_t o_wkT  = o_wqT  + (size_t)EMB * (NH * HD) * 2;
    const size_t o_wvT  = o_wkT  + (size_t)EMB * (NKV * HD) * 2;
    const size_t o_woT  = o_wvT  + (size_t)EMB * (NKV * HD) * 2;
    const size_t o_qr   = o_woT  + (size_t)(NH * HD) * EMB * 2;
    const size_t o_kr   = o_qr   + (size_t)BATCH * NH  * SEQ * HD * 2;
    const size_t o_vT   = o_kr   + (size_t)BATCH * NKV * SEQ * HD * 2;
    const size_t o_qlin = o_vT   + (size_t)BATCH * NKV * SEQ * HD * 2;
    const size_t o_klin = o_qlin + (size_t)BT * (NH * HD) * 4;
    const size_t o_vlin = o_klin + (size_t)BT * (NKV * HD) * 4;
    const size_t o_attn = o_qlin;  // alias: q_lin consumed before attn output written

    __bf16* xb   = (__bf16*)(ws + o_xb);
    __bf16* wqT  = (__bf16*)(ws + o_wqT);
    __bf16* wkT  = (__bf16*)(ws + o_wkT);
    __bf16* wvT  = (__bf16*)(ws + o_wvT);
    __bf16* woT  = (__bf16*)(ws + o_woT);
    __bf16* qr   = (__bf16*)(ws + o_qr);
    __bf16* kr   = (__bf16*)(ws + o_kr);
    __bf16* vT   = (__bf16*)(ws + o_vT);
    float*  qlin = (float*)(ws + o_qlin);
    float*  klin = (float*)(ws + o_klin);
    float*  vlin = (float*)(ws + o_vlin);
    __bf16* attn = (__bf16*)(ws + o_attn);

    auto cgrid = [](int n) { int g = (n + 255) / 256; return g > 16384 ? 16384 : g; };
    const dim3 tblk(32, 8);

    // 1) cast x to bf16; transpose-cast weights to bf16 B^T layout
    cast_f32_bf16<<<cgrid(BT * EMB), 256, 0, stream>>>(x, xb, BT * EMB);
    transpose_cast<<<dim3((NH * HD) / 32, EMB / 32), tblk, 0, stream>>>(wq, wqT, EMB, NH * HD);
    transpose_cast<<<dim3((NKV * HD) / 32, EMB / 32), tblk, 0, stream>>>(wk, wkT, EMB, NKV * HD);
    transpose_cast<<<dim3((NKV * HD) / 32, EMB / 32), tblk, 0, stream>>>(wv, wvT, EMB, NKV * HD);
    transpose_cast<<<dim3(EMB / 32, (NH * HD) / 32), tblk, 0, stream>>>(wo, woT, NH * HD, EMB);

    // 2) projections (WMMA bf16, f32 accumulate)
    gemm_bf16_f32<<<dim3((NH * HD) / 64, BT / 128), 128, 0, stream>>>(xb, wqT, qlin, BT, NH * HD, EMB);
    gemm_bf16_f32<<<dim3((NKV * HD) / 64, BT / 128), 128, 0, stream>>>(xb, wkT, klin, BT, NKV * HD, EMB);
    gemm_bf16_f32<<<dim3((NKV * HD) / 64, BT / 128), 128, 0, stream>>>(xb, wvT, vlin, BT, NKV * HD, EMB);

    // 3) RoPE (+ head-major transpose + bf16 cast); V transposed to (B,NKV,D,T)
    rope_to_bf16<<<BATCH * NH * SEQ, 64, 0, stream>>>(qlin, qr, NH);
    rope_to_bf16<<<BATCH * NKV * SEQ, 64, 0, stream>>>(klin, kr, NKV);
    v_transpose_cast<<<dim3(HD / 32, SEQ / 32, BATCH * NKV), tblk, 0, stream>>>(vlin, vT);

    // 4) causal flash attention
    flash_attn<<<dim3(SEQ / 64, NH, BATCH), 128, 0, stream>>>(qr, kr, vT, attn);

    // 5) output projection -> f32 d_out
    gemm_bf16_f32<<<dim3(EMB / 64, BT / 128), 128, 0, stream>>>(attn, woT, out, BT, EMB, NH * HD);
}
